// Speller_84052509982929
// MI455X (gfx1250) — matrix operand order repaired and checked by reading, not compiled, not run on hardware
//
#include <hip/hip_runtime.h>

#define BS   128
#define RAL  1024
#define LAL  256
#define CS   256
#define VOC  34
#define HFS  512
#define EMB  256

#define NWG    32
#define NTHR   256
#define NWAVES 8

typedef __bf16 bf16;
typedef __attribute__((ext_vector_type(16))) __bf16 v16bf;
typedef __attribute__((ext_vector_type(8)))  __bf16 v8bf;
typedef __attribute__((ext_vector_type(4)))  __bf16 v4bf;
typedef __attribute__((ext_vector_type(8)))  float  v8f;

// ---------------- shared memory layout (floats) ----------------
#define SM_TILE 0            // 8 waves * 16*17 = 2176
#define SM_HB   2176         // 4*512  = 2048
#define SM_Q    4224         // 4*256  = 1024
#define SM_E    5248         // 4*1024 = 4096
#define SM_CTX  9344         // 4*256  = 1024
#define SM_P    10368        // 256
#define SM_RED  10624        // 8
#define SM_TOT  10632

struct SpellerArgs {
  const float *key, *val, *mask, *emb;
  const float *Wih_a, *Whh_a, *bih_a, *bhh_a;
  const float *Wih_b, *Whh_b, *bih_b, *bhh_b;
  const float *Wq, *bq, *Wc, *bc;
  const int   *y;
  float *out;
  bf16 *keyB, *valB, *W1p, *W2p, *A1p, *A2p0, *A2p1;
  float *caF, *cbF, *haF, *hbF, *b1, *b2;
  unsigned *bar;   // [0]=count, [1]=generation
};

__device__ __forceinline__ float sigf(float x) { return 1.0f / (1.0f + __expf(-x)); }

// Device-wide sense-reversing barrier (all NWG blocks resident).
__device__ __forceinline__ void gridbar(unsigned *bar)
{
  __threadfence();
  __syncthreads();
  if (threadIdx.x == 0) {
    unsigned gen = __atomic_load_n(bar + 1, __ATOMIC_ACQUIRE);
    unsigned arrived = __atomic_fetch_add(bar, 1u, __ATOMIC_ACQ_REL);
    if (arrived == NWG - 1) {
      __atomic_store_n(bar, 0u, __ATOMIC_RELAXED);
      __atomic_fetch_add(bar + 1, 1u, __ATOMIC_RELEASE);
    } else {
      while (__atomic_load_n(bar + 1, __ATOMIC_ACQUIRE) == gen)
        __builtin_amdgcn_s_sleep(2);
    }
  }
  __syncthreads();
  __threadfence();
}

// One fused LSTM layer: z = [x|h_prev_layer_inputs] @ [Wih|Whh]^T  (M=128, N=2048, K=1024)
// Each wave: job = (mt, nc), computes the i/f/g/o 16x16 tiles for the same columns,
// applies gate nonlinearities in-register, updates c/h, and repacks its h tile into
// the WMMA A-fragment layout of the NEXT GEMM via an in-LDS transpose.
__device__ __forceinline__ void lstm_phase(
    const bf16 *__restrict__ Ap, const bf16 *__restrict__ Wp,
    const float *__restrict__ bias, float *__restrict__ cF, float *__restrict__ hF,
    bf16 *__restrict__ A2dst, int kt2base, float *smemTile)
{
  const int lane = threadIdx.x & 31;
  const int wv   = threadIdx.x >> 5;
  const int j    = blockIdx.x * NWAVES + wv;   // 0..255
  const int mt   = j >> 5;                     // 0..7   (16-row tile of batch)
  const int nc   = j & 31;                     // 0..31  (16-col chunk within each gate)

  v8f acc0 = {0,0,0,0,0,0,0,0};
  v8f acc1 = acc0, acc2 = acc0, acc3 = acc0;

  for (int kt = 0; kt < 32; ++kt) {
    v16bf a  = *(const v16bf *)(Ap + (((size_t)mt * 32 + kt) * 32 + lane) * 16);
    v16bf b0 = *(const v16bf *)(Wp + ((((size_t)( 0 + nc)) * 32 + kt) * 32 + lane) * 16);
    v16bf b1 = *(const v16bf *)(Wp + ((((size_t)(32 + nc)) * 32 + kt) * 32 + lane) * 16);
    v16bf b2 = *(const v16bf *)(Wp + ((((size_t)(64 + nc)) * 32 + kt) * 32 + lane) * 16);
    v16bf b3 = *(const v16bf *)(Wp + ((((size_t)(96 + nc)) * 32 + kt) * 32 + lane) * 16);
    if (kt + 2 < 32) {
      __builtin_prefetch(Ap + (((size_t)mt * 32 + kt + 2) * 32 + lane) * 16, 0, 1);
      __builtin_prefetch(Wp + ((((size_t)(0 + nc)) * 32 + kt + 2) * 32 + lane) * 16, 0, 1);
    }
    acc0 = __builtin_amdgcn_wmma_f32_16x16x32_bf16(false, a, false, b0, (short)0, acc0, false, false);
    acc1 = __builtin_amdgcn_wmma_f32_16x16x32_bf16(false, a, false, b1, (short)0, acc1, false, false);
    acc2 = __builtin_amdgcn_wmma_f32_16x16x32_bf16(false, a, false, b2, (short)0, acc2, false, false);
    acc3 = __builtin_amdgcn_wmma_f32_16x16x32_bf16(false, a, false, b3, (short)0, acc3, false, false);
  }

  const int colq  = nc * 16 + (lane & 15);       // column within one gate [0,512)
  const int rbase = mt * 16 + 8 * (lane >> 4);   // + v  (D layout: row = v + 8*(lane/16))
  const float bi = bias[colq], bff = bias[512 + colq];
  const float bg = bias[1024 + colq], bo = bias[1536 + colq];

  float *tile = smemTile + wv * (16 * 17);       // [row_local][col_local], padded
  #pragma unroll
  for (int v = 0; v < 8; ++v) {
    int r = rbase + v;
    float iv = sigf(acc0[v] + bi);
    float fv = sigf(acc1[v] + bff);
    float gv = tanhf(acc2[v] + bg);
    float ov = sigf(acc3[v] + bo);
    float c  = fv * cF[r * HFS + colq] + iv * gv;
    cF[r * HFS + colq] = c;
    float h = ov * tanhf(c);
    hF[r * HFS + colq] = h;
    tile[(v + 8 * (lane >> 4)) * 17 + (lane & 15)] = h;
  }
  __syncthreads();
  {
    // D-layout -> A-fragment layout repack of this wave's 16x16 h tile.
    const int kt2 = kt2base + (nc >> 1);
    const int hs  = nc & 1;                      // which K-half of the A tile we fill
    const int g   = lane >> 4;
    const int rl  = lane & 15;
    v8bf pk;
    #pragma unroll
    for (int q = 0; q < 8; ++q) pk[q] = (bf16)tile[rl * 17 + g * 8 + q];
    *(v8bf *)(A2dst + (((size_t)mt * 32 + kt2) * 32 + lane) * 16 + hs * 8) = pk;
  }
  __syncthreads();
}

__global__ void init_bar(unsigned *bar) { if (threadIdx.x < 2) bar[threadIdx.x] = 0u; }

__global__ __launch_bounds__(NTHR, 1) void speller_persist(SpellerArgs A)
{
  __shared__ float smem[SM_TOT];
  const int tid  = threadIdx.x;
  const int gtid = blockIdx.x * NTHR + tid;
  const int gstr = NWG * NTHR;

  // ================= prologue (runs every launch; deterministic) =================
  {  // key/val f32 -> bf16 (so attention streams from L2: 128 MB/step instead of 256)
    const float4 *k4 = (const float4 *)A.key;
    const float4 *v4 = (const float4 *)A.val;
    v4bf *kd = (v4bf *)A.keyB, *vd = (v4bf *)A.valB;
    const int n4 = BS * RAL * CS / 4;
    for (int i = gtid; i < n4; i += gstr) {
      float4 kk = k4[i], vv = v4[i];
      v4bf a, b;
      a[0]=(bf16)kk.x; a[1]=(bf16)kk.y; a[2]=(bf16)kk.z; a[3]=(bf16)kk.w;
      b[0]=(bf16)vv.x; b[1]=(bf16)vv.y; b[2]=(bf16)vv.z; b[3]=(bf16)vv.w;
      kd[i] = a; vd[i] = b;
    }
  }
  // Pack [Wih|Whh] into WMMA B-fragment layout: [nt(128)][kt(32)][lane(32)][16 bf16]
  for (int s = gtid; s < 128 * 32 * 32; s += gstr) {
    int lane = s & 31, kt = (s >> 5) & 31, nt = s >> 10;
    int col = nt * 16 + (lane & 15);
    int kb  = kt * 32 + (lane >> 4) * 8;         // B layout: lanes 0-15 K0..7,16..23; 16-31 K8..15,24..31
    v16bf w1, w2;
    #pragma unroll
    for (int e = 0; e < 16; ++e) {
      int K = kb + (e < 8 ? e : e + 8);
      float x1 = (K < 512) ? A.Wih_a[col * 512 + K] : A.Whh_a[col * 512 + K - 512];
      float x2 = (K < 512) ? A.Wih_b[col * 512 + K] : A.Whh_b[col * 512 + K - 512];
      w1[e] = (bf16)x1; w2[e] = (bf16)x2;
    }
    size_t off = (((size_t)nt * 32 + kt) * 32 + lane) * 16;
    *(v16bf *)(A.W1p + off) = w1;
    *(v16bf *)(A.W2p + off) = w2;
  }
  for (int i = gtid; i < 2048; i += gstr) {      // fused biases
    A.b1[i] = A.bih_a[i] + A.bhh_a[i];
    A.b2[i] = A.bih_b[i] + A.bhh_b[i];
  }
  for (int i = gtid; i < BS * HFS; i += gstr) { A.caF[i] = 0.f; A.cbF[i] = 0.f; }
  for (int i = gtid; i < 8 * 32 * 32 * 16 / 2; i += gstr) ((unsigned *)A.A2p0)[i] = 0u;
  // Pack A1 for t=0: x = [emb(y0) | ctx=0 | ha=0] in A-fragment layout
  for (int s = gtid; s < 8 * 32 * 32; s += gstr) {
    int lane = s & 31, kt = (s >> 5) & 31, mt = s >> 10;
    int row = mt * 16 + (lane & 15);
    int g = lane >> 4;
    int yi = A.y[row * LAL + 0];
    v16bf pk;
    #pragma unroll
    for (int e = 0; e < 16; ++e) {
      int K = kt * 32 + g * 8 + (e < 8 ? e : e + 8);
      float v = (K < EMB) ? A.emb[yi * EMB + K] : 0.f;
      pk[e] = (bf16)v;
    }
    *(v16bf *)(A.A1p + (((size_t)mt * 32 + kt) * 32 + lane) * 16) = pk;
  }
  gridbar(A.bar);

  // ================= recurrence =================
  const int r0 = blockIdx.x * 4;                 // this WG's 4 batch rows for attention
  for (int t = 0; t < LAL; ++t) {
    bf16 *A2rd = (t & 1) ? A.A2p1 : A.A2p0;      // double-buffered [ha|hb] A fragments
    bf16 *A2wr = (t & 1) ? A.A2p0 : A.A2p1;

    // ---- LSTM A: x=[emb|ctx|ha] ; writes ha, ca, and ha-half of A2rd ----
    lstm_phase(A.A1p, A.W1p, A.b1, A.caF, A.haF, A2rd, 0, smem + SM_TILE);
    gridbar(A.bar);
    // ---- LSTM B: x=[ha|hb_prev] ; writes hb, cb, and hb-half of A2wr (next step) ----
    lstm_phase(A2rd, A.W2p, A.b2, A.cbF, A.hbF, A2wr, 16, smem + SM_TILE);
    gridbar(A.bar);

    // ---- per-WG: query, attention, context, logits, pack A1(t+1) ----
    {
      float *hbL  = smem + SM_HB;
      float *qL   = smem + SM_Q;
      float *eL   = smem + SM_E;
      float *ctxL = smem + SM_CTX;
      float *pbuf = smem + SM_P;
      float *red  = smem + SM_RED;

      for (int i = tid; i < 4 * HFS; i += NTHR)
        hbL[i] = A.hbF[(size_t)(r0 + i / HFS) * HFS + (i % HFS)];
      __syncthreads();

      { // query = hb @ Wq^T + bq ; thread = one output column for all 4 rows
        int c = tid;
        float a0 = A.bq[c], a1 = a0, a2 = a0, a3 = a0;
        const float *wq = A.Wq + (size_t)c * HFS;
        for (int k = 0; k < HFS; ++k) {
          float w = wq[k];
          a0 += w * hbL[0 * HFS + k]; a1 += w * hbL[1 * HFS + k];
          a2 += w * hbL[2 * HFS + k]; a3 += w * hbL[3 * HFS + k];
        }
        qL[0 * CS + c] = a0; qL[1 * CS + c] = a1; qL[2 * CS + c] = a2; qL[3 * CS + c] = a3;
      }
      __syncthreads();

      { // energy[r][p] = key[r][p] . q[r]
        int r = tid >> 6, p0 = tid & 63;
        const bf16 *kb = A.keyB + (size_t)(r0 + r) * RAL * CS;
        for (int pp = 0; pp < 16; ++pp) {
          int p = p0 + pp * 64;
          const v8bf *krow = (const v8bf *)(kb + (size_t)p * CS);
          __builtin_prefetch(kb + (size_t)(p0 + (pp + 1) * 64) * CS, 0, 1);
          float e = 0.f;
          for (int cc = 0; cc < CS / 8; ++cc) {
            v8bf kv = krow[cc];
            #pragma unroll
            for (int q = 0; q < 8; ++q) e += (float)kv[q] * qL[r * CS + cc * 8 + q];
          }
          eL[r * RAL + p] = e;
        }
      }
      __syncthreads();
      { // masked softmax: row max
        int r = tid >> 6, p0 = tid & 63;
        float m = -1e30f;
        for (int pp = 0; pp < 16; ++pp) m = fmaxf(m, eL[r * RAL + p0 + pp * 64]);
        pbuf[tid] = m;
      }
      __syncthreads();
      if (tid < 4) {
        float m = -1e30f;
        for (int i = 0; i < 64; ++i) m = fmaxf(m, pbuf[tid * 64 + i]);
        red[tid] = m;
      }
      __syncthreads();
      { // exp * mask, partial sums
        int r = tid >> 6, p0 = tid & 63;
        float m = red[r], s = 0.f;
        const float *mk = A.mask + (size_t)(r0 + r) * RAL;
        for (int pp = 0; pp < 16; ++pp) {
          int p = p0 + pp * 64;
          float pv = __expf(eL[r * RAL + p] - m) * mk[p];
          eL[r * RAL + p] = pv; s += pv;
        }
        pbuf[tid] = s;
      }
      __syncthreads();
      if (tid < 4) {
        float s = 0.f;
        for (int i = 0; i < 64; ++i) s += pbuf[tid * 64 + i];
        red[4 + tid] = 1.f / s;
      }
      __syncthreads();
      {
        int r = tid >> 6, p0 = tid & 63;
        float inv = red[4 + r];
        for (int pp = 0; pp < 16; ++pp) eL[r * RAL + p0 + pp * 64] *= inv;
      }
      __syncthreads();
      { // context: ctx[r][c] = sum_p att[r][p] * val[r][p][c] ; coalesced bf16 stream
        int r = tid >> 6, c0 = (tid & 63) * 4;
        const bf16 *vb = A.valB + (size_t)(r0 + r) * RAL * CS + c0;
        float a0 = 0.f, a1 = 0.f, a2 = 0.f, a3 = 0.f;
        for (int p = 0; p < RAL; ++p) {
          float w = eL[r * RAL + p];
          v4bf vv = *(const v4bf *)(vb + (size_t)p * CS);
          a0 += w * (float)vv[0]; a1 += w * (float)vv[1];
          a2 += w * (float)vv[2]; a3 += w * (float)vv[3];
        }
        ctxL[r * CS + c0 + 0] = a0; ctxL[r * CS + c0 + 1] = a1;
        ctxL[r * CS + c0 + 2] = a2; ctxL[r * CS + c0 + 3] = a3;
      }
      __syncthreads();
      if (tid < 4 * VOC) { // logits = [hb|ctx] @ Wc^T + bc
        int r = tid / VOC, v = tid % VOC;
        const float *wc = A.Wc + (size_t)v * (HFS + CS);
        float acc = A.bc[v];
        for (int k = 0; k < HFS; ++k) acc += wc[k] * hbL[r * HFS + k];
        for (int k = 0; k < CS; ++k)  acc += wc[HFS + k] * ctxL[r * CS + k];
        A.out[((size_t)(r0 + r) * LAL + t) * VOC + v] = acc;
      }
      if (t + 1 < LAL) { // pack A1(t+1) = [emb(y_{t+1}) | ctx | ha] for this WG's 4 rows
        int kt = tid >> 3, rr = (tid >> 1) & 3, g = tid & 1;
        int row = r0 + rr;
        int mt = row >> 4, lane = (row & 15) + 16 * g;
        int yi = A.y[row * LAL + (t + 1)];
        v16bf pk;
        #pragma unroll
        for (int e = 0; e < 16; ++e) {
          int K = kt * 32 + g * 8 + (e < 8 ? e : e + 8);
          float v;
          if (K < EMB)            v = A.emb[yi * EMB + K];
          else if (K < EMB + CS)  v = ctxL[rr * CS + (K - EMB)];
          else                    v = A.haF[(size_t)row * HFS + (K - EMB - CS)];
          pk[e] = (bf16)v;
        }
        *(v16bf *)(A.A1p + (((size_t)mt * 32 + kt) * 32 + lane) * 16) = pk;
      }
    }
    gridbar(A.bar);
  }
}

extern "C" void kernel_launch(void *const *d_in, const int *in_sizes, int n_in,
                              void *d_out, int out_size, void *d_ws, size_t ws_size,
                              hipStream_t stream)
{
  (void)in_sizes; (void)n_in; (void)out_size; (void)ws_size;
  SpellerArgs A;
  A.key   = (const float *)d_in[0];
  A.val   = (const float *)d_in[1];
  A.mask  = (const float *)d_in[2];
  A.emb   = (const float *)d_in[3];
  A.Wih_a = (const float *)d_in[4];
  A.Whh_a = (const float *)d_in[5];
  A.bih_a = (const float *)d_in[6];
  A.bhh_a = (const float *)d_in[7];
  A.Wih_b = (const float *)d_in[8];
  A.Whh_b = (const float *)d_in[9];
  A.bih_b = (const float *)d_in[10];
  A.bhh_b = (const float *)d_in[11];
  A.Wq    = (const float *)d_in[12];
  A.bq    = (const float *)d_in[13];
  A.Wc    = (const float *)d_in[14];
  A.bc    = (const float *)d_in[15];
  A.y     = (const int *)d_in[16];
  A.out   = (float *)d_out;

  char *w = (char *)d_ws;
  auto take = [&](size_t bytes) -> char * {
    char *p = w; w += (bytes + 255) & ~size_t(255); return p;
  };
  A.keyB = (bf16 *)take((size_t)BS * RAL * CS * 2);      // 64 MB
  A.valB = (bf16 *)take((size_t)BS * RAL * CS * 2);      // 64 MB
  A.W1p  = (bf16 *)take((size_t)128 * 32 * 32 * 16 * 2); // 4 MB
  A.W2p  = (bf16 *)take((size_t)128 * 32 * 32 * 16 * 2); // 4 MB
  A.A1p  = (bf16 *)take((size_t)8 * 32 * 32 * 16 * 2);   // 256 KB
  A.A2p0 = (bf16 *)take((size_t)8 * 32 * 32 * 16 * 2);
  A.A2p1 = (bf16 *)take((size_t)8 * 32 * 32 * 16 * 2);
  A.caF  = (float *)take((size_t)BS * HFS * 4);
  A.cbF  = (float *)take((size_t)BS * HFS * 4);
  A.haF  = (float *)take((size_t)BS * HFS * 4);
  A.hbF  = (float *)take((size_t)BS * HFS * 4);
  A.b1   = (float *)take(2048 * 4);
  A.b2   = (float *)take(2048 * 4);
  A.bar  = (unsigned *)take(256);

  init_bar<<<1, 64, 0, stream>>>(A.bar);
  speller_persist<<<NWG, NTHR, 0, stream>>>(A);
}
